// PoemDecoder_80152679678772
// MI455X (gfx1250) — compile-verified, matrix operand order checked by reading, not run on hardware
//
#include <hip/hip_runtime.h>
#include <hip/hip_bf16.h>

// Problem dims
#define BB 16
#define SS 512
#define TT 128
#define VV 32000
#define EE 256
#define HH 512
#define LL 2
#define NHH 8
#define HDD 64

typedef __bf16 bf16x16 __attribute__((ext_vector_type(16)));
typedef float  f32x8   __attribute__((ext_vector_type(8)));

__device__ __forceinline__ unsigned short f2b(float f) {
    unsigned int u = __float_as_uint(f);
    unsigned int r = 0x7FFFu + ((u >> 16) & 1u);   // round-to-nearest-even
    return (unsigned short)((u + r) >> 16);
}
__device__ __forceinline__ float sigm(float x) { return 1.0f / (1.0f + __expf(-x)); }
__device__ __forceinline__ float gelu(float x) { return 0.5f * x * (1.0f + erff(x * 0.70710678118f)); }

struct FragAB { union { bf16x16 v; uint4 q[2]; }; };
struct FragC  { union { f32x8  v; float f[8]; }; };

// A-fragment (16x32 bf16) / B-fragment (32x16 bf16) per-lane load.
// lane<16 holds K {0..7, 16..23}; lane>=16 holds K {8..15, 24..31} (ISA 7.12.2).
// "row" = row m of A (X row-major) or row n of W (since B col n == W row n).
__device__ __forceinline__ void load_frag(FragAB& fr, const unsigned short* row, int k, int hi) {
    const unsigned short* p = row + k + (hi << 3);
    fr.q[0] = *(const uint4*)(p);
    fr.q[1] = *(const uint4*)(p + 16);
}

// ---------------------------------------------------------------------------
// Generic M=16 GEMM:  out[16xN] = A1[16xK1] @ W1[NxK1]^T (+ A2[16xK2] @ W2^T) + bias
// A operands in bf16 (global, contiguous 16 x K), staged into LDS.
// act: 0 = none, 1 = exact GELU.  Writes fp32 (ldout-strided) and/or bf16.
// ---------------------------------------------------------------------------
__global__ void k_gemm16(const unsigned short* __restrict__ A1, int K1,
                         const unsigned short* __restrict__ Wm1, int ldw1,
                         const unsigned short* __restrict__ A2, int K2,
                         const unsigned short* __restrict__ Wm2, int ldw2,
                         const float* __restrict__ bias,
                         float* __restrict__ outf, unsigned short* __restrict__ outb,
                         long long ldout, int N, int act) {
    __shared__ unsigned short As[16 * 1024];
    const int tid = threadIdx.x;

    { // stage A1 (+A2) into LDS as dwords
        const unsigned int* a1 = (const unsigned int*)A1;
        unsigned int* s1 = (unsigned int*)As;
        const int n1 = (16 * K1) >> 1;
        for (int i = tid; i < n1; i += 256) s1[i] = a1[i];
        if (A2) {
            const unsigned int* a2 = (const unsigned int*)A2;
            unsigned int* s2 = (unsigned int*)(As + 16 * K1);
            const int n2 = (16 * K2) >> 1;
            for (int i = tid; i < n2; i += 256) s2[i] = a2[i];
        }
    }
    __syncthreads();

    const int wave = tid >> 5, lane = tid & 31;
    const int lr = lane & 15, hi = lane >> 4;
    const int tile = blockIdx.x * 8 + wave;
    const int n0 = tile * 16;
    if (n0 >= N) return;

    FragC acc;
    const float bval = bias[n0 + lr];
#pragma unroll
    for (int r = 0; r < 8; ++r) acc.f[r] = bval;

    {
        const unsigned short* arow = As + lr * K1;
        const unsigned short* wrow = Wm1 + (size_t)(n0 + lr) * ldw1;
        for (int k = 0; k < K1; k += 32) {
            FragAB a, b;
            load_frag(a, arow, k, hi);
            load_frag(b, wrow, k, hi);
            acc.v = __builtin_amdgcn_wmma_f32_16x16x32_bf16(
                false, a.v, false, b.v, (short)0, acc.v, false, false);
        }
    }
    if (A2) {
        const unsigned short* arow = As + 16 * K1 + lr * K2;
        const unsigned short* wrow = Wm2 + (size_t)(n0 + lr) * ldw2;
        for (int k = 0; k < K2; k += 32) {
            FragAB a, b;
            load_frag(a, arow, k, hi);
            load_frag(b, wrow, k, hi);
            acc.v = __builtin_amdgcn_wmma_f32_16x16x32_bf16(
                false, a.v, false, b.v, (short)0, acc.v, false, false);
        }
    }

    if (act == 1) {
#pragma unroll
        for (int r = 0; r < 8; ++r) acc.f[r] = gelu(acc.f[r]);
    }

#pragma unroll
    for (int r = 0; r < 8; ++r) {
        const int m = r + (hi << 3);          // D row (ISA 7.12.2 C/D layout)
        const int n = n0 + lr;                // D column
        if (outf) outf[(size_t)m * (size_t)ldout + n] = acc.f[r];
        if (outb) outb[(size_t)m * (size_t)ldout + n] = f2b(acc.f[r]);
    }
}

// ---------------------------------------------------------------------------
// One-time K/V projection: kh/vh[B,S,H] = enc @ Wk^T + bk  (and Wv/bv).
// Grid: (B*S)/16 = 512 blocks; each block converts its 16 enc rows to bf16 in
// LDS, then 8 waves cover 64 output tiles (32 for K, 32 for V).
// ---------------------------------------------------------------------------
__global__ void k_kv(const float* __restrict__ enc,
                     const unsigned short* __restrict__ Wkb,
                     const unsigned short* __restrict__ Wvb,
                     const float* __restrict__ bk, const float* __restrict__ bv,
                     float* __restrict__ kh, float* __restrict__ vh) {
    __shared__ unsigned short As[16 * HH];
    const int tid = threadIdx.x;
    const int row0 = blockIdx.x * 16;  // in [0, B*S)

    for (int i = tid; i < 16 * HH; i += 256)
        As[i] = f2b(enc[(size_t)row0 * HH + i]);
    __syncthreads();

    const int wave = tid >> 5, lane = tid & 31;
    const int lr = lane & 15, hi = lane >> 4;

    for (int tile = wave; tile < 64; tile += 8) {
        const int mat = tile >> 5;
        const int n0 = (tile & 31) * 16;
        const unsigned short* W = mat ? Wvb : Wkb;
        const float* bias = mat ? bv : bk;
        float* out = mat ? vh : kh;

        FragC acc;
        const float bval = bias[n0 + lr];
#pragma unroll
        for (int r = 0; r < 8; ++r) acc.f[r] = bval;

        const unsigned short* arow = As + lr * HH;
        const unsigned short* wrow = W + (size_t)(n0 + lr) * HH;
        for (int k = 0; k < HH; k += 32) {
            FragAB a, b;
            load_frag(a, arow, k, hi);
            load_frag(b, wrow, k, hi);
            acc.v = __builtin_amdgcn_wmma_f32_16x16x32_bf16(
                false, a.v, false, b.v, (short)0, acc.v, false, false);
        }
#pragma unroll
        for (int r = 0; r < 8; ++r) {
            const int m = r + (hi << 3);
            out[(size_t)(row0 + m) * HH + n0 + lr] = acc.f[r];
        }
    }
}

// ---------------------------------------------------------------------------
// Fused single-query MHA for one (batch, head): q-proj, scores, softmax,
// weighted sum. Writes per-head softmax weights (for the attn_w output) and
// the bf16 context vector (input to the Wo GEMM).
// ---------------------------------------------------------------------------
__global__ void k_attn(const float* __restrict__ h_top,
                       const float* __restrict__ Wq, const float* __restrict__ bq,
                       const float* __restrict__ kh, const float* __restrict__ vh,
                       const int* __restrict__ mask,
                       float* __restrict__ wsw, unsigned short* __restrict__ ctxb) {
    const int b = blockIdx.x >> 3;
    const int head = blockIdx.x & 7;
    const int tid = threadIdx.x;  // 256

    __shared__ float hs[HH];
    __shared__ float qs[HDD];
    __shared__ float sc[SS];
    __shared__ float red[256];

    for (int i = tid; i < HH; i += 256) hs[i] = h_top[b * HH + i];
    __syncthreads();

    if (tid < HDD) {
        const float* wrow = Wq + (size_t)(head * HDD + tid) * HH;
        float s = bq[head * HDD + tid];
        for (int k = 0; k < HH; ++k) s += hs[k] * wrow[k];
        qs[tid] = s;
    }
    __syncthreads();

    for (int s = tid; s < SS; s += 256) {
        const float* krow = kh + (size_t)(b * SS + s) * HH + head * HDD;
        float a = 0.0f;
        for (int d = 0; d < HDD; ++d) a += qs[d] * krow[d];
        a *= 0.125f;  // 1/sqrt(64)
        if (mask[b * SS + s] == 0) a = -1e9f;
        sc[s] = a;
    }
    __syncthreads();

    float m = -3.0e38f;
    for (int s = tid; s < SS; s += 256) m = fmaxf(m, sc[s]);
    red[tid] = m; __syncthreads();
    for (int st = 128; st > 0; st >>= 1) { if (tid < st) red[tid] = fmaxf(red[tid], red[tid + st]); __syncthreads(); }
    m = red[0]; __syncthreads();

    float ssum = 0.0f;
    for (int s = tid; s < SS; s += 256) { float e = __expf(sc[s] - m); sc[s] = e; ssum += e; }
    red[tid] = ssum; __syncthreads();
    for (int st = 128; st > 0; st >>= 1) { if (tid < st) red[tid] += red[tid + st]; __syncthreads(); }
    const float inv = 1.0f / red[0]; __syncthreads();

    for (int s = tid; s < SS; s += 256) {
        const float w = sc[s] * inv;
        sc[s] = w;
        wsw[(size_t)(b * NHH + head) * SS + s] = w;
    }
    __syncthreads();

    // context: ctx[d] = sum_s w[s] * vh[b,s,head,d]
    const int d = tid & 63, part = tid >> 6;  // 4 partial chunks of 128
    float acc = 0.0f;
    for (int s = part * 128; s < part * 128 + 128; ++s)
        acc += sc[s] * vh[(size_t)(b * SS + s) * HH + head * HDD + d];
    red[tid] = acc; __syncthreads();
    if (part == 0) {
        const float v = red[d] + red[64 + d] + red[128 + d] + red[192 + d];
        ctxb[b * HH + head * HDD + d] = f2b(v);
    }
}

// mean over heads of the attention weights -> d_out attn section [B,T,S]
__global__ void k_wmean(const float* __restrict__ wsw, float* __restrict__ aw, int t) {
    const int idx = blockIdx.x * 256 + threadIdx.x;  // B*S = 8192
    if (idx >= BB * SS) return;
    const int b = idx >> 9, s = idx & 511;
    float a = 0.0f;
    for (int h = 0; h < NHH; ++h) a += wsw[(size_t)(b * NHH + h) * SS + s];
    aw[((size_t)b * TT + t) * SS + s] = a * (1.0f / NHH);
}

// embedding lookup for step t -> bf16 [16 x 256]
__global__ void k_embed(const int* __restrict__ ids, const float* __restrict__ emb,
                        unsigned short* __restrict__ xemb, int t) {
    const int idx = blockIdx.x * 256 + threadIdx.x;  // B*E = 4096
    if (idx >= BB * EE) return;
    const int b = idx >> 8, e = idx & 255;
    const int tok = ids[b * TT + t];
    xemb[b * EE + e] = f2b(emb[(size_t)tok * EE + e]);
}

// LSTM pointwise gate nonlinearity (PyTorch i,f,g,o order)
__global__ void k_point(const float* __restrict__ gates,
                        float* __restrict__ c, float* __restrict__ h,
                        unsigned short* __restrict__ hb) {
    const int idx = blockIdx.x * 256 + threadIdx.x;  // B*H = 8192
    if (idx >= BB * HH) return;
    const int b = idx >> 9, j = idx & 511;
    const float* g = gates + (size_t)b * 4 * HH;
    const float i_ = g[j], f_ = g[HH + j], gg = g[2 * HH + j], o_ = g[3 * HH + j];
    const float cn = sigm(f_) * c[idx] + sigm(i_) * tanhf(gg);
    const float hn = sigm(o_) * tanhf(cn);
    c[idx] = cn; h[idx] = hn; hb[idx] = f2b(hn);
}

// LayerNorm over H=512, emits bf16 for the W1 GEMM
__global__ void k_ln(const float* __restrict__ x, const float* __restrict__ g,
                     const float* __restrict__ be, unsigned short* __restrict__ out) {
    const int b = blockIdx.x, tid = threadIdx.x;  // 16 blocks x 256
    __shared__ float red[256];
    float s = 0.0f;
    for (int j = tid; j < HH; j += 256) s += x[b * HH + j];
    red[tid] = s; __syncthreads();
    for (int st = 128; st > 0; st >>= 1) { if (tid < st) red[tid] += red[tid + st]; __syncthreads(); }
    const float mu = red[0] * (1.0f / HH); __syncthreads();
    float v = 0.0f;
    for (int j = tid; j < HH; j += 256) { const float d = x[b * HH + j] - mu; v += d * d; }
    red[tid] = v; __syncthreads();
    for (int st = 128; st > 0; st >>= 1) { if (tid < st) red[tid] += red[tid + st]; __syncthreads(); }
    const float inv = rsqrtf(red[0] * (1.0f / HH) + 1e-5f); __syncthreads();
    for (int j = tid; j < HH; j += 256)
        out[b * HH + j] = f2b((x[b * HH + j] - mu) * inv * g[j] + be[j]);
}

__global__ void k_f2bf(const float* __restrict__ s, unsigned short* __restrict__ d, int n) {
    const int i = blockIdx.x * 256 + threadIdx.x;
    if (i < n) d[i] = f2b(s[i]);
}

__global__ void k_bsum(const float* __restrict__ a, const float* __restrict__ b,
                       float* __restrict__ d, int n) {
    const int i = blockIdx.x * 256 + threadIdx.x;
    if (i < n) d[i] = a[i] + b[i];
}

// h[0]=h[1]=context, c=0, plus bf16 copies of h
__global__ void k_init(const float* __restrict__ ctx, float* __restrict__ h,
                       float* __restrict__ c, unsigned short* __restrict__ hb) {
    const int idx = blockIdx.x * 256 + threadIdx.x;  // B*H
    if (idx >= BB * HH) return;
    const float v = ctx[idx];
    h[idx] = v; h[BB * HH + idx] = v;
    c[idx] = 0.0f; c[BB * HH + idx] = 0.0f;
    const unsigned short bv = f2b(v);
    hb[idx] = bv; hb[BB * HH + idx] = bv;
}

extern "C" void kernel_launch(void* const* d_in, const int* in_sizes, int n_in,
                              void* d_out, int out_size, void* d_ws, size_t ws_size,
                              hipStream_t stream) {
    const float* enc  = (const float*)d_in[0];
    const float* ctx  = (const float*)d_in[1];
    const int*   tgt  = (const int*)d_in[2];
    const int*   mask = (const int*)d_in[3];
    const float* emb  = (const float*)d_in[4];
    const float* Wq = (const float*)d_in[5],  *bq = (const float*)d_in[6];
    const float* Wk = (const float*)d_in[7],  *bk = (const float*)d_in[8];
    const float* Wv = (const float*)d_in[9],  *bv = (const float*)d_in[10];
    const float* Wo = (const float*)d_in[11], *bo = (const float*)d_in[12];
    const float* Wp = (const float*)d_in[13], *bp = (const float*)d_in[14];
    const float* Wih = (const float*)d_in[15], *Whh = (const float*)d_in[16];
    const float* bih = (const float*)d_in[17], *bhh = (const float*)d_in[18];
    const float* Wc = (const float*)d_in[19], *bc = (const float*)d_in[20];
    const float* lng = (const float*)d_in[21], *lnb = (const float*)d_in[22];
    const float* W1 = (const float*)d_in[23], *b1 = (const float*)d_in[24];
    const float* W2 = (const float*)d_in[25], *b2 = (const float*)d_in[26];

    // ---- workspace carve (256B aligned slots) ----
    char* wsb = (char*)d_ws;
    size_t off = 0;
    auto carve = [&](size_t bytes) -> void* {
        void* p = wsb + off;
        off = (off + bytes + 255) & ~(size_t)255;
        return p;
    };
    unsigned short* W2b  = (unsigned short*)carve((size_t)VV * HH * 2);
    unsigned short* Wpb  = (unsigned short*)carve((size_t)HH * (EE + HH) * 2);
    unsigned short* Wihb = (unsigned short*)carve((size_t)LL * 4 * HH * HH * 2);
    unsigned short* Whhb = (unsigned short*)carve((size_t)LL * 4 * HH * HH * 2);
    unsigned short* Wcb  = (unsigned short*)carve((size_t)HH * 2 * HH * 2);
    unsigned short* W1b  = (unsigned short*)carve((size_t)HH * HH * 2);
    unsigned short* Wob  = (unsigned short*)carve((size_t)HH * HH * 2);
    unsigned short* Wkb  = (unsigned short*)carve((size_t)HH * HH * 2);
    unsigned short* Wvb  = (unsigned short*)carve((size_t)HH * HH * 2);
    float* kh    = (float*)carve((size_t)BB * SS * HH * 4);
    float* vh    = (float*)carve((size_t)BB * SS * HH * 4);
    float* bsum  = (float*)carve((size_t)LL * 4 * HH * 4);
    float* hst   = (float*)carve((size_t)LL * BB * HH * 4);
    float* cst   = (float*)carve((size_t)LL * BB * HH * 4);
    unsigned short* hb   = (unsigned short*)carve((size_t)LL * BB * HH * 2);
    unsigned short* ctxb = (unsigned short*)carve((size_t)BB * HH * 2);  // pre-Wo context
    unsigned short* attb = (unsigned short*)carve((size_t)BB * HH * 2);  // post-Wo attention
    unsigned short* xemb = (unsigned short*)carve((size_t)BB * EE * 2);
    unsigned short* x0b  = (unsigned short*)carve((size_t)BB * HH * 2);
    float* gates = (float*)carve((size_t)BB * 4 * HH * 4);
    float* outc  = (float*)carve((size_t)BB * HH * 4);
    unsigned short* lnx  = (unsigned short*)carve((size_t)BB * HH * 2);
    unsigned short* t1b  = (unsigned short*)carve((size_t)BB * HH * 2);
    float* wsw   = (float*)carve((size_t)BB * NHH * SS * 4);

    float* logits = (float*)d_out;
    float* aw     = logits + (size_t)BB * TT * VV;

    // ---- one-time prep ----
    auto cvt = [&](const float* s, unsigned short* d, int n) {
        k_f2bf<<<(n + 255) / 256, 256, 0, stream>>>(s, d, n);
    };
    cvt(W2, W2b, VV * HH);
    cvt(Wp, Wpb, HH * (EE + HH));
    cvt(Wih, Wihb, LL * 4 * HH * HH);
    cvt(Whh, Whhb, LL * 4 * HH * HH);
    cvt(Wc, Wcb, HH * 2 * HH);
    cvt(W1, W1b, HH * HH);
    cvt(Wo, Wob, HH * HH);
    cvt(Wk, Wkb, HH * HH);
    cvt(Wv, Wvb, HH * HH);
    k_bsum<<<(LL * 4 * HH + 255) / 256, 256, 0, stream>>>(bih, bhh, bsum, LL * 4 * HH);
    k_init<<<(BB * HH + 255) / 256, 256, 0, stream>>>(ctx, hst, cst, hb);
    k_kv<<<BB * SS / 16, 256, 0, stream>>>(enc, Wkb, Wvb, bk, bv, kh, vh);

    const size_t Lw = (size_t)4 * HH * HH;  // per-layer weight stride (elements)
    float* h0 = hst;            float* h1 = hst + BB * HH;
    float* c0 = cst;            float* c1 = cst + BB * HH;
    unsigned short* hb0 = hb;   unsigned short* hb1 = hb + BB * HH;

    // ---- sequential decode ----
    for (int t = 0; t < TT; ++t) {
        // attention with query = top-layer hidden state (previous step)
        k_attn<<<BB * NHH, 256, 0, stream>>>(h1, Wq, bq, kh, vh, mask, wsw, ctxb);
        k_wmean<<<BB * SS / 256, 256, 0, stream>>>(wsw, aw, t);
        // attn = ctx @ Wo^T + bo  (bf16 out only)
        k_gemm16<<<HH / 128, 256, 0, stream>>>(ctxb, HH, Wob, HH,
                                               nullptr, 0, nullptr, 0, bo,
                                               nullptr, attb, HH, HH, 0);
        // inp = [emb ; attn] @ Wp^T + bp
        k_embed<<<BB * EE / 256, 256, 0, stream>>>(tgt, emb, xemb, t);
        k_gemm16<<<HH / 128, 256, 0, stream>>>(xemb, EE, Wpb, EE + HH,
                                               attb, HH, Wpb + EE, EE + HH, bp,
                                               nullptr, x0b, HH, HH, 0);
        // LSTM layer 0
        k_gemm16<<<4 * HH / 128, 256, 0, stream>>>(x0b, HH, Wihb, HH,
                                                   hb0, HH, Whhb, HH, bsum,
                                                   gates, nullptr, 4 * HH, 4 * HH, 0);
        k_point<<<BB * HH / 256, 256, 0, stream>>>(gates, c0, h0, hb0);
        // LSTM layer 1
        k_gemm16<<<4 * HH / 128, 256, 0, stream>>>(hb0, HH, Wihb + Lw, HH,
                                                   hb1, HH, Whhb + Lw, HH, bsum + 4 * HH,
                                                   gates, nullptr, 4 * HH, 4 * HH, 0);
        k_point<<<BB * HH / 256, 256, 0, stream>>>(gates, c1, h1, hb1);
        // out = LN([h_top ; attn] @ Wc^T + bc)
        k_gemm16<<<HH / 128, 256, 0, stream>>>(hb1, HH, Wcb, 2 * HH,
                                               attb, HH, Wcb + HH, 2 * HH, bc,
                                               outc, nullptr, HH, HH, 0);
        k_ln<<<BB, 256, 0, stream>>>(outc, lng, lnb, lnx);
        // t1 = gelu(out @ W1^T + b1)
        k_gemm16<<<HH / 128, 256, 0, stream>>>(lnx, HH, W1b, HH,
                                               nullptr, 0, nullptr, 0, b1,
                                               nullptr, t1b, HH, HH, 1);
        // logits[:, t, :] = t1 @ W2^T + b2   (W2 bf16 stays L2-resident)
        k_gemm16<<<VV / 128, 256, 0, stream>>>(t1b, HH, W2b, HH,
                                               nullptr, 0, nullptr, 0, b2,
                                               logits + (size_t)t * VV, nullptr,
                                               (long long)TT * VV, VV, 0);
    }
}
